// Head_19645180411975
// MI455X (gfx1250) — compile-verified, matrix operand order checked by reading, not run on hardware
//
#include <hip/hip_runtime.h>
#include <hip/hip_bf16.h>

// ---------------------------------------------------------------------------
// Fused causal attention head (B=4, T=2048, C=1024) for gfx1250 (MI455X).
// bf16 WMMA (V_WMMA_F32_16X16X32_BF16), register-blocked 4x4 (64x64) tiles.
// __launch_bounds__(256,1) lifts the VGPR cap so the 16x v8f accumulators
// (128 VGPRs) + 8 fragments (64 VGPRs) live in registers without spilling.
// ---------------------------------------------------------------------------

typedef __attribute__((ext_vector_type(16))) __bf16 v16bf;
typedef __attribute__((ext_vector_type(8)))  __bf16 v8bf;
typedef __attribute__((ext_vector_type(8)))  float  v8f;

#define B_  4
#define T_  2048
#define C_  1024

// Fragment loader for 16x32 bf16 tiles, row-major source (ISA 7.12.2 layout).
// rowptr = &M[row_of_this_lane][k0]; kb = 8*(lane>>4).
// Lane holds K = [kb, kb+8) and [16+kb, 16+kb+8).
// The B-fragment is the identical mapping applied to rows of B^T.
__device__ __forceinline__ v16bf load_frag(const __bf16* rowptr, int kb) {
  v8bf lo = *(const v8bf*)(rowptr + kb);
  v8bf hi = *(const v8bf*)(rowptr + kb + 16);
  return __builtin_shufflevector(lo, hi, 0,1,2,3,4,5,6,7,8,9,10,11,12,13,14,15);
}

__device__ __forceinline__ v8f wmma_bf16(v16bf a, v16bf b, v8f c) {
  return __builtin_amdgcn_wmma_f32_16x16x32_bf16(false, a, false, b,
                                                 (short)0, c, false, false);
}

// ---------------------------------------------------------------------------
// Kernel 1: fp32 -> bf16 conversion; fold C^-0.5 = 1/32 into Wq.
// ---------------------------------------------------------------------------
__global__ void convert_kernel(const float* __restrict__ x,
                               const float* __restrict__ Wk,
                               const float* __restrict__ Wq,
                               const float* __restrict__ Wv,
                               __bf16* __restrict__ xb,
                               __bf16* __restrict__ wkb,
                               __bf16* __restrict__ wqb,
                               __bf16* __restrict__ wvb) {
  size_t i = (size_t)blockIdx.x * blockDim.x + threadIdx.x;
  const size_t NX = (size_t)B_ * T_ * C_;
  const size_t NW = (size_t)C_ * C_;
  if (i < NX) xb[i] = (__bf16)x[i];
  if (i < NW) {
    wkb[i] = (__bf16)Wk[i];
    wqb[i] = (__bf16)(Wq[i] * 0.03125f);   // scale folded into Q projection
    wvb[i] = (__bf16)Wv[i];
  }
}

// ---------------------------------------------------------------------------
// Kernel 2: projections Q = x*Wq^T (pre-scaled), K = x*Wk^T, V = x*Wv^T.
// One wave -> one 64x64 output tile (4x4 WMMAs per K-chunk).
// blockDim = 256 (8 waves). grid = (C/64, (B*T)/64/8, 3).
// ---------------------------------------------------------------------------
__global__ void __launch_bounds__(256, 1)
proj_kernel(const __bf16* __restrict__ xb,
            const __bf16* __restrict__ wqb,
            const __bf16* __restrict__ wkb,
            const __bf16* __restrict__ wvb,
            __bf16* __restrict__ qb,
            __bf16* __restrict__ kbuf,
            __bf16* __restrict__ vtb) {
  const int wave = threadIdx.x >> 5;
  const int lane = threadIdx.x & 31;
  const int mi = blockIdx.y * 8 + wave;   // 0..127  (64-row tile over B*T)
  const int ni = blockIdx.x;              // 0..15   (64-col tile over C)
  const int which = blockIdx.z;           // 0=Q, 1=K, 2=V

  const __bf16* W = (which == 0) ? wqb : (which == 1) ? wkb : wvb;
  const int m  = lane & 15;
  const int hi = lane >> 4;
  const int kb = hi * 8;

  const __bf16* A  = xb + (size_t)(mi * 64) * C_;
  const __bf16* Bm = W  + (size_t)(ni * 64) * C_;

  v8f acc[4][4] = {};
  for (int k0 = 0; k0 < C_; k0 += 32) {
    v16bf af[4], bfr[4];
    #pragma unroll
    for (int i = 0; i < 4; ++i)
      af[i] = load_frag(A + (size_t)(i * 16 + m) * C_ + k0, kb);
    #pragma unroll
    for (int j = 0; j < 4; ++j)
      bfr[j] = load_frag(Bm + (size_t)(j * 16 + m) * C_ + k0, kb);
    #pragma unroll
    for (int i = 0; i < 4; ++i)
      #pragma unroll
      for (int j = 0; j < 4; ++j)
        acc[i][j] = wmma_bf16(af[i], bfr[j], acc[i][j]);
  }

  const int n   = lane & 15;
  const int gm0 = mi * 64;
  if (which == 2) {
    const int bb = gm0 / T_;              // 64 | 2048: tiles never straddle a batch
    #pragma unroll
    for (int i = 0; i < 4; ++i)
      #pragma unroll
      for (int j = 0; j < 4; ++j)
        #pragma unroll
        for (int r = 0; r < 8; ++r) {
          int row = gm0 + i * 16 + r + 8 * hi;
          int t   = row & (T_ - 1);
          int gn  = ni * 64 + j * 16 + n;
          vtb[((size_t)bb * C_ + gn) * T_ + t] = (__bf16)acc[i][j][r];
        }
  } else {
    __bf16* dst = (which == 0) ? qb : kbuf;
    #pragma unroll
    for (int i = 0; i < 4; ++i)
      #pragma unroll
      for (int j = 0; j < 4; ++j)
        #pragma unroll
        for (int r = 0; r < 8; ++r) {
          int row = gm0 + i * 16 + r + 8 * hi;
          int gn  = ni * 64 + j * 16 + n;
          dst[(size_t)row * C_ + gn] = (__bf16)acc[i][j][r];
        }
  }
}

// ---------------------------------------------------------------------------
// Kernel 3: S = Q * K^T (scale folded into Q). 64x64 tiles, lower-triangle
// blocks only (si <= ti at 64-granularity; diagonal blocks include masked
// elements that softmax ignores). blockDim = 256. grid = (T/64/8, T/64, B).
// ---------------------------------------------------------------------------
__global__ void __launch_bounds__(256, 1)
score_kernel(const __bf16* __restrict__ qb,
             const __bf16* __restrict__ kbuf,
             __bf16* __restrict__ wei) {
  const int wave = threadIdx.x >> 5;
  const int lane = threadIdx.x & 31;
  const int si = blockIdx.x * 8 + wave;   // 64-wide key block 0..31
  const int ti = blockIdx.y;              // 64-tall query block 0..31
  const int b  = blockIdx.z;
  if (si > ti) return;                    // wave-uniform: EXEC all-1 for WMMA

  const int m  = lane & 15;
  const int hi = lane >> 4;
  const int kb = hi * 8;
  const __bf16* A  = qb   + ((size_t)b * T_ + ti * 64) * C_;
  const __bf16* Bm = kbuf + ((size_t)b * T_ + si * 64) * C_;

  v8f acc[4][4] = {};
  for (int k0 = 0; k0 < C_; k0 += 32) {
    v16bf af[4], bfr[4];
    #pragma unroll
    for (int i = 0; i < 4; ++i)
      af[i] = load_frag(A + (size_t)(i * 16 + m) * C_ + k0, kb);
    #pragma unroll
    for (int j = 0; j < 4; ++j)
      bfr[j] = load_frag(Bm + (size_t)(j * 16 + m) * C_ + k0, kb);
    #pragma unroll
    for (int i = 0; i < 4; ++i)
      #pragma unroll
      for (int j = 0; j < 4; ++j)
        acc[i][j] = wmma_bf16(af[i], bfr[j], acc[i][j]);
  }

  const int n = lane & 15;
  #pragma unroll
  for (int i = 0; i < 4; ++i)
    #pragma unroll
    for (int j = 0; j < 4; ++j)
      #pragma unroll
      for (int r = 0; r < 8; ++r) {
        int t = ti * 64 + i * 16 + r + 8 * hi;
        int s = si * 64 + j * 16 + n;
        wei[((size_t)b * T_ + t) * T_ + s] = (__bf16)acc[i][j][r];
      }
}

// ---------------------------------------------------------------------------
// Kernel 4: causal softmax, in place on the bf16 score buffer.
// One wave per row; zeros written for s > t so kernel 5 can round K up to 32.
// blockDim = 256 (8 rows/block). grid = B*T/8.
// ---------------------------------------------------------------------------
__global__ void softmax_kernel(__bf16* __restrict__ wei) {
  const int wave = threadIdx.x >> 5;
  const int lane = threadIdx.x & 31;
  const int row  = blockIdx.x * 8 + wave;   // 0..B*T-1
  const int b = row >> 11;
  const int t = row & (T_ - 1);
  __bf16* w = wei + ((size_t)b * T_ + t) * T_;

  float m = -3.0e38f;
  for (int s = lane; s <= t; s += 32) m = fmaxf(m, (float)w[s]);
  #pragma unroll
  for (int off = 16; off; off >>= 1) m = fmaxf(m, __shfl_xor(m, off, 32));

  float sum = 0.0f;
  for (int s = lane; s <= t; s += 32) sum += __expf((float)w[s] - m);
  #pragma unroll
  for (int off = 16; off; off >>= 1) sum += __shfl_xor(sum, off, 32);
  const float inv = 1.0f / sum;

  for (int s = lane; s < T_; s += 32) {
    float p = (s <= t) ? __expf((float)w[s] - m) * inv : 0.0f;
    w[s] = (__bf16)p;
  }
}

// ---------------------------------------------------------------------------
// Kernel 5: out = P * V using V^T (contiguous B-fragment loads).
// One wave -> 32x64 tile (2x4 WMMAs per K-chunk); causal K extent = ti+1
// chunks of 32 (wei zero-filled above the diagonal).
// blockDim = 256. grid = (C/64, T/32/8, B).
// ---------------------------------------------------------------------------
__global__ void __launch_bounds__(256, 1)
out_kernel(const __bf16* __restrict__ wei,
           const __bf16* __restrict__ vtb,
           float* __restrict__ out) {
  const int wave = threadIdx.x >> 5;
  const int lane = threadIdx.x & 31;
  const int ti = blockIdx.y * 8 + wave;   // 32-row query tile 0..63
  const int dn = blockIdx.x;              // 64-wide channel tile 0..15
  const int b  = blockIdx.z;

  const int m  = lane & 15;
  const int hi = lane >> 4;
  const int kb = hi * 8;
  const __bf16* A  = wei + ((size_t)b * T_ + ti * 32) * T_;
  const __bf16* Bm = vtb + ((size_t)b * C_ + dn * 64) * T_;

  v8f acc[2][4] = {};
  const int nch = ti + 1;                 // ceil((ti*32+32)/32)
  for (int c = 0; c < nch; ++c) {
    int k0 = c * 32;
    v16bf af[2], bfr[4];
    #pragma unroll
    for (int i = 0; i < 2; ++i)
      af[i] = load_frag(A + (size_t)(i * 16 + m) * T_ + k0, kb);
    #pragma unroll
    for (int j = 0; j < 4; ++j)
      bfr[j] = load_frag(Bm + (size_t)(j * 16 + m) * T_ + k0, kb);
    #pragma unroll
    for (int i = 0; i < 2; ++i)
      #pragma unroll
      for (int j = 0; j < 4; ++j)
        acc[i][j] = wmma_bf16(af[i], bfr[j], acc[i][j]);
  }

  const int n = lane & 15;
  #pragma unroll
  for (int i = 0; i < 2; ++i)
    #pragma unroll
    for (int j = 0; j < 4; ++j)
      #pragma unroll
      for (int r = 0; r < 8; ++r) {
        int t = ti * 32 + i * 16 + r + 8 * hi;
        int d = dn * 64 + j * 16 + n;
        out[((size_t)b * T_ + t) * C_ + d] = acc[i][j][r];
      }
}

// ---------------------------------------------------------------------------
// Launcher. setup_inputs order: x, Wk, Wq, Wv. Output fp32 [B,T,C].
// Workspace layout (bytes):
//   xb   @ 0          16 MiB   x as bf16
//   wkb  @ 16777216    2 MiB
//   wqb  @ 18874368    2 MiB   (pre-scaled by C^-0.5)
//   wvb  @ 20971520    2 MiB
//   qb   @ 23068672   16 MiB
//   kbuf @ 39845888   16 MiB
//   vtb  @ 56623104   16 MiB   V transposed [b][d][t]
//   wei  @ 73400320   32 MiB   bf16 scores -> probabilities (in place)
//   total 106,954,752 bytes
// ---------------------------------------------------------------------------
extern "C" void kernel_launch(void* const* d_in, const int* in_sizes, int n_in,
                              void* d_out, int out_size, void* d_ws, size_t ws_size,
                              hipStream_t stream) {
  const float* x  = (const float*)d_in[0];
  const float* Wk = (const float*)d_in[1];
  const float* Wq = (const float*)d_in[2];
  const float* Wv = (const float*)d_in[3];
  float* out = (float*)d_out;

  char* ws = (char*)d_ws;
  __bf16* xb   = (__bf16*)(ws + 0);
  __bf16* wkb  = (__bf16*)(ws + 16777216);
  __bf16* wqb  = (__bf16*)(ws + 18874368);
  __bf16* wvb  = (__bf16*)(ws + 20971520);
  __bf16* qb   = (__bf16*)(ws + 23068672);
  __bf16* kbuf = (__bf16*)(ws + 39845888);
  __bf16* vtb  = (__bf16*)(ws + 56623104);
  __bf16* wei  = (__bf16*)(ws + 73400320);

  // 1) convert to bf16
  convert_kernel<<<dim3(32768), dim3(256), 0, stream>>>(x, Wk, Wq, Wv,
                                                        xb, wkb, wqb, wvb);
  // 2) Q/K/V projections (V transposed), 64x64 tiles per wave
  proj_kernel<<<dim3(16, 16, 3), dim3(256), 0, stream>>>(xb, wqb, wkb, wvb,
                                                         qb, kbuf, vtb);
  // 3) causal score blocks, 64x64 tiles per wave
  score_kernel<<<dim3(4, 32, B_), dim3(256), 0, stream>>>(qb, kbuf, wei);
  // 4) row softmax (in place, zero-fills upper triangle)
  softmax_kernel<<<dim3(1024), dim3(256), 0, stream>>>(wei);
  // 5) out = P * V, 32x64 tiles per wave
  out_kernel<<<dim3(16, 8, B_), dim3(256), 0, stream>>>(wei, vtb, out);
}
